// DecoderBlock_82420422410637
// MI455X (gfx1250) — matrix operand order repaired and checked by reading, not compiled, not run on hardware
//
#include <hip/hip_runtime.h>
#include <hip/hip_bf16.h>

#define NUM_SECTORS 11

typedef __attribute__((ext_vector_type(16))) _Float16 v16h;
typedef __attribute__((ext_vector_type(8)))  _Float16 v8h;
typedef __attribute__((ext_vector_type(8)))  float    v8f;

// ---------------------------------------------------------------------------
// WMMA helpers (CDNA5 wave32, 16x16x32 f16 -> f32), layouts per ISA 7.12.2
// ---------------------------------------------------------------------------
__device__ __forceinline__ v8f wmma16(v16h a, v16h b, v8f c) {
    return __builtin_amdgcn_wmma_f32_16x16x32_f16(
        /*neg_a=*/false, a, /*neg_b=*/false, b,
        /*c_mod=*/(short)0, c, /*reuse_a=*/false, /*reuse_b=*/false);
}

// A-matrix 16x32 f16 fragment from a row-major LDS tile (pitch in halfs).
// Lane L: m = L&15, half = L>>4. halves[0..7] = K (half*8 .. half*8+7),
// halves[8..15] = K (16+half*8 .. 16+half*8+7).  (ISA 16-bit A table)
__device__ __forceinline__ v16h load_a_frag(const _Float16* tile, int pitch, int kbase) {
    int lane = threadIdx.x & 31;
    int m = lane & 15, hf = lane >> 4;
    const _Float16* p0 = tile + m * pitch + kbase + hf * 8;
    v8h lo = *(const v8h*)p0;
    v8h hi = *(const v8h*)(p0 + 16);
    v16h a;
#pragma unroll
    for (int i = 0; i < 8; ++i) { a[i] = lo[i]; a[8 + i] = hi[i]; }
    return a;
}

// B-matrix 32x16 f16 fragment. Tile stored n-major / k-contiguous
// (tileT[n][k]).  Lane L: n = L&15, halves[0..7] = K (half*8 .. half*8+7),
// halves[8..15] = K (16+half*8 .. 16+half*8+7).
__device__ __forceinline__ v16h load_b_frag(const _Float16* tileT, int pitch, int kbase) {
    int lane = threadIdx.x & 31;
    int n = lane & 15, hf = lane >> 4;
    const _Float16* p0 = tileT + n * pitch + kbase + hf * 8;
    v8h lo = *(const v8h*)p0;
    v8h hi = *(const v8h*)(p0 + 16);
    v16h b;
#pragma unroll
    for (int i = 0; i < 8; ++i) { b[i] = lo[i]; b[8 + i] = hi[i]; }
    return b;
}

// ---------------------------------------------------------------------------
// gfx1250 async global->LDS copy (16B per lane).  Inline asm per
// CDNA5 ISA (portable across ROCm7.2 / amdgpu-toolchain).
// LDS byte address = low 32 bits of the generic pointer (ISA 10.2 aperture).
// ---------------------------------------------------------------------------
__device__ __forceinline__ void async_copy_b128(void* lds_dst, const void* gsrc) {
    unsigned lds = (unsigned)(uintptr_t)lds_dst;
    unsigned long long ga = (unsigned long long)(uintptr_t)gsrc;
    asm volatile("global_load_async_to_lds_b128 %0, %1, off"
                 :: "v"(lds), "v"(ga)
                 : "memory");
}
__device__ __forceinline__ void wait_async0() {
    asm volatile("s_wait_asynccnt 0" ::: "memory");
}

// ---------------------------------------------------------------------------
// Elementwise f32 -> f16 cast (weights)
// ---------------------------------------------------------------------------
__global__ void cast_f32_to_f16(const float* __restrict__ in,
                                _Float16* __restrict__ out, int n) {
    int i = blockIdx.x * 256 + threadIdx.x;
    if (i < n) out[i] = (_Float16)in[i];
}

// ---------------------------------------------------------------------------
// LayerNorm (C=768) + cast to f16, one wave per row
// ---------------------------------------------------------------------------
__global__ void __launch_bounds__(256) ln_cast_kernel(
    const float* __restrict__ x, const float* __restrict__ w,
    const float* __restrict__ b, _Float16* __restrict__ out) {
    const int C = 768;
    int wave = threadIdx.x >> 5, lane = threadIdx.x & 31;
    long row = (long)blockIdx.x * 8 + wave;
    const float* xr = x + row * C;
    float vals[24], s = 0.f, ss = 0.f;
#pragma unroll
    for (int i = 0; i < 24; ++i) {
        float v = xr[lane + i * 32];
        vals[i] = v; s += v; ss += v * v;
    }
#pragma unroll
    for (int off = 16; off > 0; off >>= 1) {
        s  += __shfl_xor(s,  off, 32);
        ss += __shfl_xor(ss, off, 32);
    }
    float mu = s * (1.f / C);
    float var = ss * (1.f / C) - mu * mu;
    float rs = rsqrtf(var + 1e-5f);
#pragma unroll
    for (int i = 0; i < 24; ++i) {
        int c = lane + i * 32;
        out[row * C + c] = (_Float16)((vals[i] - mu) * rs * w[c] + b[c]);
    }
}

// ---------------------------------------------------------------------------
// GEMM: C[M,N] = A[M,K](f16) * B[K,N](f16), fp32 accum via WMMA.
// Block tile 64x128, 8 waves of 32x32.  A tile staged via async global->LDS;
// B tile staged manually (transposed to n-major/k-contiguous).
// EPI=0: store f16. EPI=1: out = resid + gamma[col]*(acc + bias[col]).
// ---------------------------------------------------------------------------
template <int EPI>
__global__ void __launch_bounds__(256) gemm_h16(
    const _Float16* __restrict__ A, const _Float16* __restrict__ Bw,
    int M, int N, int K,
    _Float16* __restrict__ outH,
    const float* __restrict__ resid, const float* __restrict__ bias,
    const float* __restrict__ gamma, float* __restrict__ outF) {
    __shared__ __align__(16) _Float16 As[64 * 40];
    __shared__ __align__(16) _Float16 Bt[128 * 40];

    const int t = threadIdx.x;
    const int wave = t >> 5, lane = t & 31;
    const int m0 = blockIdx.y * 64;
    const int n0 = blockIdx.x * 128;
    const int wm = (wave >> 2) * 32;   // 0 or 32
    const int wn = (wave & 3) * 32;    // 0,32,64,96

    v8f acc[2][2];
#pragma unroll
    for (int i = 0; i < 2; ++i)
#pragma unroll
        for (int j = 0; j < 2; ++j)
#pragma unroll
            for (int r = 0; r < 8; ++r) acc[i][j][r] = 0.f;

    for (int kk = 0; kk < K; kk += 32) {
        {   // A tile 64x32, row-major: async DMA straight into LDS
            int r = t >> 2, c = (t & 3) * 8;
            async_copy_b128(&As[r * 40 + c], &A[(long)(m0 + r) * K + kk + c]);
        }
        {   // B tile 32x128 -> transposed LDS Bt[n][k]
            int kr = t >> 4;          // 0..15
            int nc = (t & 15) * 8;    // 0..120
#pragma unroll
            for (int p = 0; p < 2; ++p) {
                v8h vb = *(const v8h*)&Bw[(long)(kk + kr + p * 16) * N + n0 + nc];
#pragma unroll
                for (int i = 0; i < 8; ++i) Bt[(nc + i) * 40 + kr + p * 16] = vb[i];
            }
        }
        wait_async0();
        __syncthreads();
        v16h a0 = load_a_frag(&As[(wm +  0) * 40], 40, 0);
        v16h a1 = load_a_frag(&As[(wm + 16) * 40], 40, 0);
        v16h b0 = load_b_frag(&Bt[(wn +  0) * 40], 40, 0);
        v16h b1 = load_b_frag(&Bt[(wn + 16) * 40], 40, 0);
        acc[0][0] = wmma16(a0, b0, acc[0][0]);
        acc[0][1] = wmma16(a0, b1, acc[0][1]);
        acc[1][0] = wmma16(a1, b0, acc[1][0]);
        acc[1][1] = wmma16(a1, b1, acc[1][1]);
        __syncthreads();
    }

    // C/D layout: VGPR r -> row r + 8*(lane>>4); col = lane&15
    const int cn = lane & 15, hf = lane >> 4;
#pragma unroll
    for (int i = 0; i < 2; ++i)
#pragma unroll
        for (int j = 0; j < 2; ++j)
#pragma unroll
            for (int r = 0; r < 8; ++r) {
                long row = m0 + wm + i * 16 + r + hf * 8;
                int  col = n0 + wn + j * 16 + cn;
                float v = acc[i][j][r];
                if (EPI == 0) {
                    outH[row * N + col] = (_Float16)v;
                } else {
                    outF[row * N + col] =
                        resid[row * N + col] + gamma[col] * (v + bias[col]);
                }
            }
}

// ---------------------------------------------------------------------------
// Per-(b,h) sector means of V: vpos[s][b][h][d] = mean over tokens in sector
// ---------------------------------------------------------------------------
__global__ void __launch_bounds__(256) sector_means_kernel(
    const _Float16* __restrict__ qkv,   // (B*N, 2304) f16
    const int* __restrict__ sector_ids,
    float* __restrict__ vpos) {
    const int B = 4, N = 1024, H = 12;
    int b = blockIdx.x / H, h = blockIdx.x % H;
    __shared__ float sums[NUM_SECTORS * 64];
    __shared__ float cnts[NUM_SECTORS];
    int t = threadIdx.x;
    for (int i = t; i < NUM_SECTORS * 64; i += 256) sums[i] = 0.f;
    if (t < NUM_SECTORS) cnts[t] = 0.f;
    __syncthreads();
    int d = t & 63;
    for (int n0 = 0; n0 < N; n0 += 4) {
        int n = n0 + (t >> 6);
        float v = (float)qkv[((long)b * N + n) * 2304 + 1536 + h * 64 + d];
        int s = sector_ids[n];
        atomicAdd(&sums[s * 64 + d], v);
        if (d == 0) atomicAdd(&cnts[s], 1.0f);
    }
    __syncthreads();
    for (int i = t; i < NUM_SECTORS * 64; i += 256) {
        int s = i >> 6;
        vpos[(((long)s * B + b) * H + h) * 64 + (i & 63)] =
            sums[i] / fmaxf(cnts[s], 1.0f);
    }
}

// ---------------------------------------------------------------------------
// Flash attention per (b,h): 128 query rows/block, 16 rows/wave, key tiles
// of 64, WMMA for QK^T and PV, online softmax, gated sector-mean epilogue.
// Q and K tiles staged via async global->LDS; V^T staged manually.
// ---------------------------------------------------------------------------
__global__ void __launch_bounds__(256) attention_kernel(
    const _Float16* __restrict__ qkv,       // (B*N, 2304) f16
    const int* __restrict__ sector_ids,
    const float* __restrict__ gate_logit,
    const float* __restrict__ vpos,         // (S,B,H,64)
    _Float16* __restrict__ attnH) {         // (B*N, 768) f16
    const int B = 4, N = 1024, H = 12;
    const int bh = blockIdx.x;
    const int b = bh / H, h = bh % H;
    const int qblk = blockIdx.y;            // 0..7 (128 rows each)
    const int t = threadIdx.x, wave = t >> 5, lane = t & 31;
    const int cn = lane & 15, hf = lane >> 4;

    __shared__ __align__(16) _Float16 Qs[128 * 72];
    __shared__ __align__(16) _Float16 Ks[64 * 72];
    __shared__ __align__(16) _Float16 Vt[64 * 72];
    __shared__ __align__(16) _Float16 Ps[8][16 * 72];

    const long rowbase = (long)b * N + qblk * 128;

    {   // Q tile (128 x 64): 4 async B128 copies per thread
        int r = t >> 1, seg = (t & 1) * 32;
        const _Float16* src = &qkv[(rowbase + r) * 2304 + h * 64 + seg];
#pragma unroll
        for (int i = 0; i < 4; ++i)
            async_copy_b128(&Qs[r * 72 + seg + i * 8], &src[i * 8]);
    }

    float mrow[8], lrow[8];
    v8f o[4];
#pragma unroll
    for (int r = 0; r < 8; ++r) { mrow[r] = -1e30f; lrow[r] = 0.f; }
#pragma unroll
    for (int dt = 0; dt < 4; ++dt)
#pragma unroll
        for (int r = 0; r < 8; ++r) o[dt][r] = 0.f;

    const float scale = 0.125f;             // D=64 -> 1/sqrt(64)
    _Float16* P = &Ps[wave][0];

    for (int kt = 0; kt < N / 64; ++kt) {
        {   // K tile row-major [key][d] via async; V tile transposed [d][key]
            int r = t >> 2, seg = (t & 3) * 16;
            long krow = (long)b * N + kt * 64 + r;
            const _Float16* ksrc = &qkv[krow * 2304 + 768 + h * 64 + seg];
            async_copy_b128(&Ks[r * 72 + seg],     &ksrc[0]);
            async_copy_b128(&Ks[r * 72 + seg + 8], &ksrc[8]);
            const _Float16* vsrc = &qkv[krow * 2304 + 1536 + h * 64 + seg];
            v8h v0 = *(const v8h*)&vsrc[0];
            v8h v1 = *(const v8h*)&vsrc[8];
#pragma unroll
            for (int i = 0; i < 8; ++i) {
                Vt[(seg + i) * 72 + r]     = v0[i];
                Vt[(seg + 8 + i) * 72 + r] = v1[i];
            }
        }
        wait_async0();          // covers Q (first iter) + K tile
        __syncthreads();

        // ---- S = Q @ K^T : 16 x 64 per wave ----
        v8f s[4];
#pragma unroll
        for (int nt = 0; nt < 4; ++nt)
#pragma unroll
            for (int r = 0; r < 8; ++r) s[nt][r] = 0.f;
#pragma unroll
        for (int ks = 0; ks < 64; ks += 32) {
            v16h a = load_a_frag(&Qs[(wave * 16) * 72], 72, ks);
#pragma unroll
            for (int nt = 0; nt < 4; ++nt) {
                v16h bfr = load_b_frag(&Ks[(nt * 16) * 72], 72, ks);
                s[nt] = wmma16(a, bfr, s[nt]);
            }
        }

        // ---- online softmax; P written to per-wave LDS in A-readable form
#pragma unroll
        for (int r = 0; r < 8; ++r) {
            float tm = -1e30f;
#pragma unroll
            for (int nt = 0; nt < 4; ++nt) tm = fmaxf(tm, s[nt][r]);
#pragma unroll
            for (int off = 1; off < 16; off <<= 1)
                tm = fmaxf(tm, __shfl_xor(tm, off, 32));
            tm *= scale;
            float nm = fmaxf(mrow[r], tm);
            float corr = __expf(mrow[r] - nm);
            mrow[r] = nm;
            lrow[r] *= corr;
#pragma unroll
            for (int dt = 0; dt < 4; ++dt) o[dt][r] *= corr;
            float rs = 0.f;
#pragma unroll
            for (int nt = 0; nt < 4; ++nt) {
                float p = __expf(s[nt][r] * scale - nm);
                rs += p;
                P[(r + hf * 8) * 72 + nt * 16 + cn] = (_Float16)p;
            }
#pragma unroll
            for (int off = 1; off < 16; off <<= 1)
                rs += __shfl_xor(rs, off, 32);
            lrow[r] += rs;
        }
        __syncthreads();

        // ---- O += P @ V ----
#pragma unroll
        for (int ks = 0; ks < 64; ks += 32) {
            v16h a = load_a_frag(P, 72, ks);
#pragma unroll
            for (int dt = 0; dt < 4; ++dt) {
                v16h bfr = load_b_frag(&Vt[(dt * 16) * 72], 72, ks);
                o[dt] = wmma16(a, bfr, o[dt]);
            }
        }
        __syncthreads();
    }

    // ---- epilogue: normalize, gate with sector means, write f16 ----
    float g = 1.f / (1.f + __expf(-gate_logit[h]));
#pragma unroll
    for (int r = 0; r < 8; ++r) {
        int nl = qblk * 128 + wave * 16 + r + hf * 8;   // token index
        int sct = sector_ids[nl];
        float inv_l = 1.f / lrow[r];
#pragma unroll
        for (int dt = 0; dt < 4; ++dt) {
            int d = dt * 16 + cn;
            float vc = o[dt][r] * inv_l;
            float vp = vpos[(((long)sct * B + b) * H + h) * 64 + d];
            attnH[((long)b * N + nl) * 768 + h * 64 + d] =
                (_Float16)(g * vp + (1.f - g) * vc);
        }
    }
}

// ---------------------------------------------------------------------------
// Host launcher
// ---------------------------------------------------------------------------
extern "C" void kernel_launch(void* const* d_in, const int* in_sizes, int n_in,
                              void* d_out, int out_size, void* d_ws, size_t ws_size,
                              hipStream_t stream) {
    (void)in_sizes; (void)n_in; (void)out_size; (void)ws_size;
    const int B = 4, N = 1024, C = 768, H = 12;
    const int BN = B * N;             // 4096
    const int N3 = 3 * C;             // 2304

    const float* x          = (const float*)d_in[0];
    const int*   sector_ids = (const int*)  d_in[1];
    const float* qkv_w      = (const float*)d_in[2];
    const float* proj_w     = (const float*)d_in[3];
    const float* proj_b     = (const float*)d_in[4];
    const float* gate_logit = (const float*)d_in[5];
    const float* norm1_w    = (const float*)d_in[6];
    const float* norm1_b    = (const float*)d_in[7];
    const float* ls1_gamma  = (const float*)d_in[8];
    // d_in[9..14]: norm2/ff params -- dead code in the reference, skipped.
    float* out = (float*)d_out;

    char* ws = (char*)d_ws;
    size_t off = 0;
    _Float16* h16 = (_Float16*)(ws + off);  off += (size_t)BN * C * 2;       // 6.0 MB
    _Float16* qw16 = (_Float16*)(ws + off); off += (size_t)C * N3 * 2;       // 3.4 MB
    _Float16* pw16 = (_Float16*)(ws + off); off += (size_t)C * C * 2;        // 1.1 MB
    _Float16* qkv16 = (_Float16*)(ws + off); off += (size_t)BN * N3 * 2;     // 18 MB
    float* vpos = (float*)(ws + off);       off += (size_t)NUM_SECTORS * B * H * 64 * 4;
    _Float16* attnH = (_Float16*)(ws + off); off += (size_t)BN * C * 2;      // 6.0 MB

    // 1) weight casts
    {
        int n1 = C * N3;
        cast_f32_to_f16<<<(n1 + 255) / 256, 256, 0, stream>>>(qkv_w, qw16, n1);
        int n2 = C * C;
        cast_f32_to_f16<<<(n2 + 255) / 256, 256, 0, stream>>>(proj_w, pw16, n2);
    }
    // 2) layernorm + cast
    ln_cast_kernel<<<BN / 8, 256, 0, stream>>>(x, norm1_w, norm1_b, h16);
    // 3) qkv GEMM: (4096x768)@(768x2304) -> f16
    gemm_h16<0><<<dim3(N3 / 128, BN / 64), 256, 0, stream>>>(
        h16, qw16, BN, N3, C, qkv16, nullptr, nullptr, nullptr, nullptr);
    // 4) sector means of V
    sector_means_kernel<<<B * H, 256, 0, stream>>>(qkv16, sector_ids, vpos);
    // 5) attention (flash) + gate
    attention_kernel<<<dim3(B * H, N / 128), 256, 0, stream>>>(
        qkv16, sector_ids, gate_logit, vpos, attnH);
    // 6) proj GEMM fused with bias + layer-scale + residual -> d_out (f32)
    gemm_h16<1><<<dim3(C / 128, BN / 64), 256, 0, stream>>>(
        attnH, pw16, BN, C, C, nullptr, x, proj_b, ls1_gamma, out);
}
